// GridToMeshEncoder_69621419868948
// MI455X (gfx1250) — matrix operand order repaired and checked by reading, not compiled, not run on hardware
//
#include <hip/hip_runtime.h>

// ---------------------------------------------------------------------------
// GridToMeshEncoder for MI455X (gfx1250, wave32, WMMA).
//
// Pipeline per block (64 mesh rows x one batch):
//   gather/bilinear-interp -> f16 A tile in LDS (64x104, K padded 68->96,
//                             stride padded for bank-conflict-free ds_load)
//   GEMM1: A(64x96) x W1(96x256), +b1, relu -> f16 H tile in LDS (64x264 str)
//   GEMM2: H(64x256) x W2(256x256), +b2     -> f32 out
// All matmuls are v_wmma_f32_16x16x32_f16 (f32 accumulate), zero-init C
// (lowers to inline SRC2=0), bias added in epilogue.
//
// Wave mapping: 16 waves; wave w owns ONE 16-col N-subtile (col = w*16+n) and
// all four 16-row M-subtiles: the B fragment is loaded once per K-step from
// packed L2-resident weights (32B/lane) and reused across 4 WMMAs. A/B
// fragments for step ks+1 are double-buffered so ds/vmem latency overlaps
// the 4 independent-accumulator WMMAs of step ks.
//
// LDS strides: sA 104 halves (52 dw, 52%64 step -> 16 lanes on distinct
// 4-dword banks), sH 264 halves (132 dw, step 4) -> ds_load_b128 sweeps all
// 64 banks conflict-free.
// ---------------------------------------------------------------------------

#define N_MESH    40962
#define GRIDPTS   (721 * 1440)
#define C_IN      64
#define HID       256
#define OUTC      256
#define TILE_M    64
#define NTHREADS  512        // 16 waves of 32
#define SA_STR    104        // padded LDS stride for A tile (halves)
#define SH_STR    264        // padded LDS stride for H tile (halves)

typedef __attribute__((ext_vector_type(16))) _Float16    v16h;
typedef __attribute__((ext_vector_type(8)))  _Float16    v8h;
typedef __attribute__((ext_vector_type(8)))  float       v8f;
typedef __attribute__((ext_vector_type(4)))  float       f32x4;
typedef __attribute__((ext_vector_type(4)))  int         i32x4;
typedef __attribute__((ext_vector_type(4)))  unsigned int u32x4;

union Frag16 { v16h v; u32x4 q[2]; };
union Half8  { v8h  v; u32x4 q;    };

// ---------------------------------------------------------------------------
// Pack an f32 weight matrix (Krows x 256) into f16 B-fragment layout:
//   packed[(kt*256 + col)*32 + j] = W[kt*32 + j][col]   (0 if kt*32+j >= Krows)
// j == k_local, the contiguous k order this code uses for A and B (lane half h -> j in
// [16h,16h+15], VGPR v -> pair j=16h+2v): a lane's fragment is 32 contiguous
// bytes at (kt*256+col)*32 + 16h halves.
// ---------------------------------------------------------------------------
__global__ void pack_weights_f16(const float* __restrict__ W,
                                 _Float16* __restrict__ Wp,
                                 int Krows, int ktiles) {
    int tid = blockIdx.x * blockDim.x + threadIdx.x;
    int total = ktiles * 256 * 32;
    if (tid >= total) return;
    int j   = tid & 31;
    int col = (tid >> 5) & 255;
    int kt  = tid >> 13;             // / (256*32)
    int k   = kt * 32 + j;
    float v = (k < Krows) ? W[k * 256 + col] : 0.0f;
    Wp[tid] = (_Float16)v;
}

__device__ __forceinline__ v16h load_a_frag(const _Float16* p) {
    // p = A row base + kOff + 16h; lane half h needs K = {16h..16h+15}
    Frag16 f;
    f.q[0] = *(const u32x4*)(p);
    f.q[1] = *(const u32x4*)(p + 8);
    return f.v;
}

__device__ __forceinline__ v16h load_b_frag(const _Float16* p) {
    // p = packed weights + ((ks*256 + col)<<5) + (h<<4): 32B contiguous/lane
    Frag16 f;
    f.q[0] = ((const u32x4*)p)[0];
    f.q[1] = ((const u32x4*)p)[1];
    return f.v;
}

__global__ __launch_bounds__(NTHREADS) void encoder_kernel(
    const float* __restrict__ grid,      // (2, GRIDPTS, 64) f32
    const float* __restrict__ meshf,     // (N_MESH, 4) f32
    const int*   __restrict__ indices,   // (N_MESH, 4) i32
    const float* __restrict__ weights,   // (N_MESH, 4) f32
    const _Float16* __restrict__ w1p,    // packed (3, 256, 32) f16
    const float* __restrict__ b1,        // (256,)
    const _Float16* __restrict__ w2p,    // packed (8, 256, 32) f16
    const float* __restrict__ b2,        // (256,)
    float* __restrict__ out)             // (2, N_MESH, 256) f32
{
    __shared__ alignas(16) _Float16 sA[TILE_M * SA_STR];   // 13 KB
    __shared__ alignas(16) _Float16 sH[TILE_M * SH_STR];   // 33 KB

    const int tid   = threadIdx.x;
    const int batch = blockIdx.y;
    const int row0  = blockIdx.x * TILE_M;

    // Warm packed weights into cache (global_prefetch_b8); shared by every
    // block, trivially resident in 192MB L2 / WGP$.
    if (tid < 64) {
        __builtin_prefetch(w2p + tid * 1024, 0, 3);
        if (tid < 24) __builtin_prefetch(w1p + tid * 1024, 0, 3);
    }

    // ---------------- Phase 1: gather + bilinear interp -> sA (f16) --------
    {
        const int r = tid >> 3;        // mesh row within tile: 0..63
        const int t = tid & 7;         // 8-channel block: 0..7
        const int row = row0 + r;
        f32x4 accL = (f32x4)0.0f, accH = (f32x4)0.0f;
        if (row < N_MESH) {
            const i32x4 iv = *(const i32x4*)(indices + row * 4);
            const f32x4 wv = *(const f32x4*)(weights + row * 4);
            const float* gb = grid + (size_t)batch * GRIDPTS * C_IN;
            int   idx[4] = { iv.x, iv.y, iv.z, iv.w };
            float wgt[4] = { wv.x, wv.y, wv.z, wv.w };
#pragma unroll
            for (int c = 0; c < 4; ++c) {
                const f32x4* p = (const f32x4*)(gb + (size_t)idx[c] * C_IN + t * 8);
                accL += wgt[c] * p[0];
                accH += wgt[c] * p[1];
            }
        }
        Half8 o;
        o.v[0] = (_Float16)accL.x; o.v[1] = (_Float16)accL.y;
        o.v[2] = (_Float16)accL.z; o.v[3] = (_Float16)accL.w;
        o.v[4] = (_Float16)accH.x; o.v[5] = (_Float16)accH.y;
        o.v[6] = (_Float16)accH.z; o.v[7] = (_Float16)accH.w;
        *(u32x4*)(sA + r * SA_STR + t * 8) = o.q;

        // cols 64..95: 4 mesh features + zero padding (K pad to 96)
        for (int i = tid; i < TILE_M * 32; i += NTHREADS) {
            int rr = i >> 5;
            int c  = 64 + (i & 31);
            int rw = row0 + rr;
            float v = 0.0f;
            if (c < 68 && rw < N_MESH) v = meshf[rw * 4 + (c - 64)];
            sA[rr * SA_STR + c] = (_Float16)v;
        }
    }
    __syncthreads();

    // Wave -> (1 N-subtile, all 4 M-subtiles) mapping
    const int w    = tid >> 5;           // 0..15 : N subtile index
    const int lane = tid & 31;
    const int n    = lane & 15;          // also A-fragment row m
    const int h    = lane >> 4;
    const int col  = w * 16 + n;

    // ---------------- Phase 2: GEMM1 (K=96) + bias + relu -> sH ------------
    {
        v8f acc[4] = {};                 // inline SRC2=0 on first WMMA
        const _Float16* bp = w1p + (((size_t)col) << 5) + (h << 4);
        const _Float16* ar[4];
#pragma unroll
        for (int mi = 0; mi < 4; ++mi)
            ar[mi] = sA + (mi * 16 + n) * SA_STR + 16 * h;

        v16h bC = load_b_frag(bp);
        v16h aC[4], aN[4], bN;
#pragma unroll
        for (int mi = 0; mi < 4; ++mi) aC[mi] = load_a_frag(ar[mi]);

#pragma unroll
        for (int ks = 0; ks < 3; ++ks) {
            if (ks < 2) {                // prefetch next K-step fragments
                bN = load_b_frag(bp + ((size_t)(ks + 1) << 13));
#pragma unroll
                for (int mi = 0; mi < 4; ++mi)
                    aN[mi] = load_a_frag(ar[mi] + (ks + 1) * 32);
            }
#pragma unroll
            for (int mi = 0; mi < 4; ++mi)
                acc[mi] = __builtin_amdgcn_wmma_f32_16x16x32_f16(
                    false, aC[mi], false, bC, (short)0, acc[mi], false, false);
            if (ks < 2) {
#pragma unroll
                for (int mi = 0; mi < 4; ++mi) aC[mi] = aN[mi];
                bC = bN;
            }
        }
        const float b1v = b1[col];
#pragma unroll
        for (int mi = 0; mi < 4; ++mi)
#pragma unroll
            for (int r = 0; r < 8; ++r) {   // C layout: VGPR r -> M = r + 8h
                float v = acc[mi][r] + b1v;
                sH[(mi * 16 + r + 8 * h) * SH_STR + col] =
                    (_Float16)(v > 0.0f ? v : 0.0f);
            }
    }
    __syncthreads();

    // ---------------- Phase 3: GEMM2 (K=256) + bias -> global --------------
    {
        v8f acc[4] = {};
        const _Float16* bp = w2p + (((size_t)col) << 5) + (h << 4);
        const _Float16* ar[4];
#pragma unroll
        for (int mi = 0; mi < 4; ++mi)
            ar[mi] = sH + (mi * 16 + n) * SH_STR + 16 * h;

        v16h bC = load_b_frag(bp);
        v16h aC[4], aN[4], bN;
#pragma unroll
        for (int mi = 0; mi < 4; ++mi) aC[mi] = load_a_frag(ar[mi]);

#pragma unroll
        for (int ks = 0; ks < 8; ++ks) {
            if (ks < 7) {                // prefetch next K-step fragments
                bN = load_b_frag(bp + ((size_t)(ks + 1) << 13));
#pragma unroll
                for (int mi = 0; mi < 4; ++mi)
                    aN[mi] = load_a_frag(ar[mi] + (ks + 1) * 32);
            }
#pragma unroll
            for (int mi = 0; mi < 4; ++mi)
                acc[mi] = __builtin_amdgcn_wmma_f32_16x16x32_f16(
                    false, aC[mi], false, bC, (short)0, acc[mi], false, false);
            if (ks < 7) {
#pragma unroll
                for (int mi = 0; mi < 4; ++mi) aC[mi] = aN[mi];
                bC = bN;
            }
        }
        const float b2v = b2[col];
        float* ob = out + ((size_t)batch * N_MESH) * OUTC;
#pragma unroll
        for (int mi = 0; mi < 4; ++mi)
#pragma unroll
            for (int r = 0; r < 8; ++r) {
                int row = row0 + mi * 16 + r + 8 * h;
                if (row < N_MESH) ob[(size_t)row * OUTC + col] = acc[mi][r] + b2v;
            }
    }
}

extern "C" void kernel_launch(void* const* d_in, const int* in_sizes, int n_in,
                              void* d_out, int out_size, void* d_ws, size_t ws_size,
                              hipStream_t stream) {
    const float* grid    = (const float*)d_in[0];
    const float* meshf   = (const float*)d_in[1];
    const int*   indices = (const int*)d_in[2];
    const float* weights = (const float*)d_in[3];
    const float* W1      = (const float*)d_in[4];
    const float* b1      = (const float*)d_in[5];
    const float* W2      = (const float*)d_in[6];
    const float* b2      = (const float*)d_in[7];
    float*       outp    = (float*)d_out;

    _Float16* w1p = (_Float16*)d_ws;                               // 3*256*32 halves
    _Float16* w2p = (_Float16*)((char*)d_ws + (size_t)3 * 256 * 32 * sizeof(_Float16));

    const int n1 = 3 * 256 * 32;
    const int n2 = 8 * 256 * 32;
    pack_weights_f16<<<(n1 + 255) / 256, 256, 0, stream>>>(W1, w1p, 68, 3);
    pack_weights_f16<<<(n2 + 255) / 256, 256, 0, stream>>>(W2, w2p, 256, 8);

    dim3 gridDim3((N_MESH + TILE_M - 1) / TILE_M, 2);
    encoder_kernel<<<gridDim3, NTHREADS, 0, stream>>>(
        grid, meshf, indices, weights, w1p, b1, w2p, b2, outp);
}
